// MetricLogits_20100446945292
// MI455X (gfx1250) — matrix operand order repaired and checked by reading, not compiled, not run on hardware
//
#include <hip/hip_runtime.h>

// Problem dims (compile-time constants from the reference)
#define BATCH   4096
#define CLASSES 50000
#define DIM     1024

typedef __attribute__((ext_vector_type(16))) __bf16 v16bf;
typedef __attribute__((ext_vector_type(8)))  float  v8f;

// 32-byte WMMA operand fragment viewed as two 16B LDS loads
union Frag {
    v16bf v;
    uint4 q[2];
};

// LDS tile geometry: per buffer, A = 128 rows x 64 k, B = 128 cols x 64 k,
// rows padded to 72 elements (144B -> 36-bank stride, conflict-free b128 reads).
#define KW     64                       // staged k-window
#define LROW   72                       // elements per padded LDS row
#define LTILE  (128 * LROW)             // 9216 elements per tile
#define LBUF   (2 * LTILE)              // A + B per buffer = 18432 elements
#define NSTEPS (DIM / KW)               // 16

// fp32 -> bf16 round-to-nearest-even (inputs are finite; no NaN path needed)
__device__ __forceinline__ unsigned short f2bf(float x) {
    unsigned int u = __float_as_uint(x);
    u += 0x7fffu + ((u >> 16) & 1u);
    return (unsigned short)(u >> 16);
}

__device__ __forceinline__ v8f wmma_bf16(const v16bf& a, const v16bf& b, const v8f& c) {
    return __builtin_amdgcn_wmma_f32_16x16x32_bf16(false, a, false, b,
                                                   (short)0, c, false, false);
}

// Async DMA: 16B global -> LDS, tracked by ASYNCcnt (ISA 15.18.3 op 98).
// INST_OFFSET is added to BOTH the global address and the LDS address
// so the caller passes
// VDST = lds_target - ioff and keeps the global base register fixed.
__device__ __forceinline__ void async_b128(unsigned lds_minus_off, const void* gbase, int ioff) {
    asm volatile("global_load_async_to_lds_b128 %0, %1, off offset:%2"
                 :: "v"(lds_minus_off), "v"(gbase), "i"(ioff) : "memory");
}

__device__ __forceinline__ void wait_async0() {
    asm volatile("s_wait_asynccnt 0x0" ::: "memory");
}

__global__ void ml_zero_sum(double* s) { *s = 0.0; }

// One block per row: fp32 row -> bf16 row + row squared-norm.
__global__ __launch_bounds__(256) void ml_convert(const float* __restrict__ src,
                                                  unsigned short* __restrict__ dst,
                                                  float* __restrict__ rowsq) {
    const int row = blockIdx.x;
    const int t   = threadIdx.x;              // 256 threads * 4 elems = 1024 = DIM
    const float4 v = ((const float4*)(src + (size_t)row * DIM))[t];
    ushort4 o;
    o.x = f2bf(v.x); o.y = f2bf(v.y); o.z = f2bf(v.z); o.w = f2bf(v.w);
    ((ushort4*)(dst + (size_t)row * DIM))[t] = o;

    __shared__ float red[256];
    red[t] = v.x * v.x + v.y * v.y + v.z * v.z + v.w * v.w;
    __syncthreads();
    for (int s = 128; s > 0; s >>= 1) {
        if (t < s) red[t] += red[t + s];
        __syncthreads();
    }
    if (t == 0) rowsq[row] = red[0];
}

// Block tile 128(M) x 128(N); 8 waves, each wave 32(M) x 64(N).
// Pipeline: async global->LDS staging (double-buffered, ASYNCcnt) overlapped
// with ds_load_b128 fragment reads + v_wmma_f32_16x16x32_bf16.
// Step loop fully unrolled so accumulator C/D operands coalesce in place.
__global__ __launch_bounds__(256) void ml_gemm(const unsigned short* __restrict__ fbf,
                                               const unsigned short* __restrict__ wbf,
                                               const float* __restrict__ f2,
                                               const float* __restrict__ w2,
                                               float* __restrict__ out_val,
                                               double* __restrict__ gsum) {
    __shared__ unsigned short smem[2 * LBUF];            // 72 KB

    const int tid   = threadIdx.x;
    const int lane  = tid & 31;
    const int wave  = tid >> 5;
    const int waveM = wave & 3;               // 4 waves along M
    const int waveN = wave >> 2;              // 2 waves along N
    const int m_blk  = blockIdx.x * 128;
    const int n_blk  = blockIdx.y * 128;
    const int m_base = m_blk + waveM * 32;
    const int n_base = n_blk + waveN * 64;
    const int nloc   = lane & 15;
    const int khalf  = lane >> 4;             // lane group 0/1

    // ---- staging addresses (per thread: 64B of A + 64B of B per step) ----
    const int srow  = tid >> 1;               // 0..127
    const int shalf = tid & 1;                // 32-element half of the 64-k window
    const unsigned sbase = (unsigned)(size_t)(&smem[0]);          // LDS byte offset
    int ccol = n_blk + srow;                                      // clamp edge cols
    ccol = (ccol < CLASSES) ? ccol : 0;
    const unsigned short* gA = fbf + (size_t)(m_blk + srow) * DIM + shalf * 32;
    const unsigned short* gB = wbf + (size_t)ccol * DIM + shalf * 32;
    const unsigned lA = sbase + (unsigned)(srow * 2 * LROW + shalf * 64);
    const unsigned lB = sbase + (unsigned)(2 * LTILE + srow * 2 * LROW + shalf * 64);

    // ---- fragment read offsets within one LDS buffer (elements) ----
    // A (16-bit 16x32): lane group selects 16-element K-half (matches B)
    const int abase = (waveM * 32 + nloc) * LROW + khalf * 16;
    // B (dense 32x16): lane group selects 16-element K-half
    const int bbase = LTILE + (waveN * 64 + nloc) * LROW + khalf * 16;

    bool valid[4];
#pragma unroll
    for (int t = 0; t < 4; ++t)
        valid[t] = (n_base + t * 16 + 16) <= CLASSES;   // CLASSES % 16 == 0

    v8f acc[2][4] = {};

    // Prologue: stage k-window 0 into buffer 0 (8 x 16B per thread)
#pragma unroll
    for (int q = 0; q < 4; ++q) {
        async_b128(lA, gA, q * 16);
        async_b128(lB, gB, q * 16);
    }
    wait_async0();
    __syncthreads();

#pragma unroll
    for (int s = 0; s < NSTEPS; ++s) {
        // Stage next k-window into the other buffer (overlaps with compute).
        // Global base regs stay fixed; k advance rides in the offset immediate,
        // LDS side compensated via VDST = lds_target - k2 (wrap-safe mod 2^32).
        if (s + 1 < NSTEPS) {
            const int k2 = (s + 1) * KW * 2;                             // bytes
            const unsigned lofs = (unsigned)(((s + 1) & 1) * LBUF * 2);  // bytes
            const unsigned vA = lA + lofs - (unsigned)k2;
            const unsigned vB = lB + lofs - (unsigned)k2;
#pragma unroll
            for (int q = 0; q < 4; ++q) {
                async_b128(vA, gA, k2 + q * 16);
                async_b128(vB, gB, k2 + q * 16);
            }
        }

        // Compute current buffer: 2 sub-windows x (12 ds_load_b128 + 8 WMMA)
        const unsigned short* sb = smem + (s & 1) * LBUF;
#pragma unroll
        for (int j = 0; j < 2; ++j) {
            const int aj = abase + j * 32;
            const int bj = bbase + j * 32;
            Frag a0, a1, b0, b1, b2, b3;
            a0.q[0] = *(const uint4*)(sb + aj);
            a0.q[1] = *(const uint4*)(sb + aj + 8);
            a1.q[0] = *(const uint4*)(sb + aj + 16 * LROW);
            a1.q[1] = *(const uint4*)(sb + aj + 16 * LROW + 8);
            b0.q[0] = *(const uint4*)(sb + bj);
            b0.q[1] = *(const uint4*)(sb + bj + 8);
            b1.q[0] = *(const uint4*)(sb + bj + 16 * LROW);
            b1.q[1] = *(const uint4*)(sb + bj + 16 * LROW + 8);
            b2.q[0] = *(const uint4*)(sb + bj + 32 * LROW);
            b2.q[1] = *(const uint4*)(sb + bj + 32 * LROW + 8);
            b3.q[0] = *(const uint4*)(sb + bj + 48 * LROW);
            b3.q[1] = *(const uint4*)(sb + bj + 48 * LROW + 8);

            acc[0][0] = wmma_bf16(a0.v, b0.v, acc[0][0]);
            acc[1][0] = wmma_bf16(a1.v, b0.v, acc[1][0]);
            acc[0][1] = wmma_bf16(a0.v, b1.v, acc[0][1]);
            acc[1][1] = wmma_bf16(a1.v, b1.v, acc[1][1]);
            acc[0][2] = wmma_bf16(a0.v, b2.v, acc[0][2]);
            acc[1][2] = wmma_bf16(a1.v, b2.v, acc[1][2]);
            acc[0][3] = wmma_bf16(a0.v, b3.v, acc[0][3]);
            acc[1][3] = wmma_bf16(a1.v, b3.v, acc[1][3]);
        }

        // Next buffer must be fully staged (all waves) before the next step
        wait_async0();
        __syncthreads();
    }

    // Epilogue: valuation = 2*cross - ||f||^2 - ||w||^2  (= -metric)
    const int mhalf = khalf * 8;              // C/D layout: lanes 16-31 -> rows 8..15
    float f2v[2][8];
#pragma unroll
    for (int mt = 0; mt < 2; ++mt)
#pragma unroll
        for (int v = 0; v < 8; ++v)
            f2v[mt][v] = f2[m_base + mt * 16 + mhalf + v];

    float lsum = 0.f;
#pragma unroll
    for (int t = 0; t < 4; ++t) {
        if (!valid[t]) continue;
        const int c = n_base + t * 16 + nloc;
        const float w2c = w2[c];
#pragma unroll
        for (int mt = 0; mt < 2; ++mt) {
            const int m = m_base + mt * 16 + mhalf;
            float* op = out_val + (size_t)m * CLASSES + c;
#pragma unroll
            for (int v = 0; v < 8; ++v) {
                const float val = 2.f * acc[mt][t][v] - f2v[mt][v] - w2c;
                op[(size_t)v * CLASSES] = val;
                lsum += val;
            }
        }
    }

    // Block-level sum of valuation (sum(metric) = -sum(valuation))
    for (int off = 16; off > 0; off >>= 1) lsum += __shfl_down(lsum, off, 32);
    __shared__ float wsum[8];
    if (lane == 0) wsum[wave] = lsum;
    __syncthreads();
    if (tid == 0) {
        float b = 0.f;
#pragma unroll
        for (int i = 0; i < 8; ++i) b += wsum[i];
        atomicAdd(gsum, (double)b);           // global_atomic_add_f64, 12512 total
    }
}

// train = valuation + mean(metric); mean(metric) = -sum(valuation)/N
__global__ __launch_bounds__(256) void ml_finalize(const float4* __restrict__ in,
                                                   float4* __restrict__ outp,
                                                   const double* __restrict__ gsum) {
    const size_t idx = (size_t)blockIdx.x * blockDim.x + threadIdx.x;
    const float mean_metric =
        (float)(-(*gsum) / (double)((size_t)BATCH * (size_t)CLASSES));
    float4 v = in[idx];
    v.x += mean_metric; v.y += mean_metric; v.z += mean_metric; v.w += mean_metric;
    outp[idx] = v;
}

extern "C" void kernel_launch(void* const* d_in, const int* in_sizes, int n_in,
                              void* d_out, int out_size, void* d_ws, size_t ws_size,
                              hipStream_t stream) {
    const float* feat    = (const float*)d_in[0];
    // d_in[1] = label (unused by the reference outputs)
    const float* weights = (const float*)d_in[2];

    float* out       = (float*)d_out;
    float* out_val   = out;                                        // [B, C]
    float* out_train = out + (size_t)BATCH * CLASSES;              // [B, C]
    float* out_w     = out + 2 * (size_t)BATCH * CLASSES;          // [C, D]

    // Workspace layout (~111 MB): bf16 feat | bf16 weights | f2 | w2 | f64 sum
    char* ws = (char*)d_ws;
    unsigned short* fbf = (unsigned short*)ws;                               // 8 MiB
    unsigned short* wbf = (unsigned short*)(ws + (size_t)8388608);           // 102.4 MB
    float* f2 = (float*)(ws + (size_t)8388608 + (size_t)102400000);
    float* w2 = f2 + BATCH;
    double* gsum = (double*)(w2 + CLASSES);                                  // 8B-aligned

    ml_zero_sum<<<1, 1, 0, stream>>>(gsum);
    ml_convert<<<BATCH,   256, 0, stream>>>(feat,    fbf, f2);
    ml_convert<<<CLASSES, 256, 0, stream>>>(weights, wbf, w2);

    dim3 grid(BATCH / 128, (CLASSES + 127) / 128);                 // 32 x 391
    ml_gemm<<<grid, 256, 0, stream>>>(fbf, wbf, f2, w2, out_val, gsum);

    const size_t nvec4 = (size_t)BATCH * CLASSES / 4;              // 51,200,000
    ml_finalize<<<(unsigned)(nvec4 / 256), 256, 0, stream>>>(
        (const float4*)out_val, (float4*)out_train, gsum);

    hipMemcpyAsync(out_w, weights, (size_t)CLASSES * DIM * sizeof(float),
                   hipMemcpyDeviceToDevice, stream);
}
